// CMDenNet_11596411699697
// MI455X (gfx1250) — compile-verified, matrix operand order checked by reading, not run on hardware
//
#include <hip/hip_runtime.h>

// ---------------- constants from the reference ----------------
#define BQ 4        // batch
#define NBQ 16      // blocks per batch
#define PPBQ 512    // points per block
#define KNNQ 32
#define HQ 128
#define GQ 512
#define RQ 64
#define BITSQ 6
#define DIQ 256
#define DSQ 16
#define DTRQ 8
#define KCQ 4

typedef _Float16 f16x8  __attribute__((ext_vector_type(8)));
typedef _Float16 f16x16 __attribute__((ext_vector_type(16)));
typedef float    f32x8  __attribute__((ext_vector_type(8)));
typedef unsigned int u32x4 __attribute__((ext_vector_type(4)));
typedef int      i32x4 __attribute__((ext_vector_type(4)));
typedef int      i32x8 __attribute__((ext_vector_type(8)));

__device__ __forceinline__ float act_apply(float v, int act) {
    if (act == 1) { // tanh-approx gelu (jax default)
        float u = 0.7978845608028654f * (v + 0.044715f * v * v * v);
        return 0.5f * v * (1.f + tanhf(u));
    }
    if (act == 2) { return v / (1.f + __expf(-v)); }             // silu
    if (act == 3) { return (v > 20.f) ? v : log1pf(__expf(v)); } // softplus
    return v;
}

// ---------------- TDM: 2D strided tile (global f16 -> LDS), ISA 08 Sec.8 D# ----------------
__device__ __forceinline__ void tdm_load_tile(unsigned lds_off, const _Float16* g,
                                              int halves_per_row, int rows, int stride_halves)
{
    unsigned long long ga = (unsigned long long)(size_t)g;
    u32x4 g0;
    g0[0] = 1u;                                        // count=1 (valid), user mode
    g0[1] = lds_off;                                   // lds_addr (bytes)
    g0[2] = (unsigned)(ga & 0xFFFFFFFFull);            // global_addr[31:0]
    g0[3] = (unsigned)((ga >> 32) & 0x1FFFFFFull)      // global_addr[56:32]
          | 0x80000000u;                               // type=2 ("image")
    i32x8 g1;
    g1[0] = (1 << 16);                                 // workgroup_mask=0, data_size=1 (2B)
    g1[1] = (halves_per_row & 0xFFFF) << 16;           // tensor_dim0[15:0]
    g1[2] = ((unsigned)halves_per_row >> 16)           // tensor_dim0[31:16]
          | ((rows & 0xFFFF) << 16);                   // tensor_dim1[15:0]
    g1[3] = ((unsigned)rows >> 16)                     // tensor_dim1[31:16]
          | ((halves_per_row & 0xFFFF) << 16);         // tile_dim0
    g1[4] = (rows & 0xFFFF);                           // tile_dim1 (tile_dim2=0)
    g1[5] = stride_halves;                             // tensor_dim0_stride[31:0]
    g1[6] = 0;                                         // stride[47:32]=0, dim1_stride lo=0
    g1[7] = 0;
    i32x4 z4 = {0, 0, 0, 0};
#if defined(__clang_major__) && __clang_major__ >= 23
    i32x8 z8 = {0, 0, 0, 0, 0, 0, 0, 0};
    __builtin_amdgcn_tensor_load_to_lds(g0, g1, z4, z4, z8, 0);
#else
    __builtin_amdgcn_tensor_load_to_lds(g0, g1, z4, z4, 0);
#endif
}

// ---------------- f32 -> f16 pre-conversion (padded, so GEMM hot loop is guard-free) ----
__global__ void cmden_cvtA(const float* __restrict__ A, int lda, _Float16* __restrict__ Ah,
                           int M, int K, int Kp)
{
    long long t = (long long)blockIdx.x * blockDim.x + threadIdx.x;
    long long r = t / Kp, k = t % Kp;
    if (r >= 0 && t < (long long)((M + 63) & ~63) * Kp)
        Ah[t] = (r < M && k < K) ? (_Float16)A[(size_t)r * lda + k] : (_Float16)0.f;
}

__global__ void cmden_cvtW(const float* __restrict__ W, _Float16* __restrict__ Wt,
                           int K, int N, int Kp, int Np)
{
    long long t = (long long)blockIdx.x * blockDim.x + threadIdx.x;
    if (t >= (long long)Np * Kp) return;
    int n = (int)(t / Kp), k = (int)(t % Kp);
    Wt[t] = (n < N && k < K) ? (_Float16)W[(size_t)k * N + n] : (_Float16)0.f;
}

// ---------------- WMMA GEMM: C = act(Ah@Wt^T + bias) (+res) ----------------
// Ah: (Mp,Kp) f16 row-major. Wt: (Np,Kp) f16 (W transposed). Block = 4 waves = 64x64 tile;
// each wave: 16 rows x 64 cols = 4 accumulators, 4 wmma / k-step. B panel staged via TDM.
__global__ __launch_bounds__(128)
void cmden_gemm_f16(const _Float16* __restrict__ Ah, int Kp,
                    const _Float16* __restrict__ Wt,
                    const float* __restrict__ bias,
                    float* __restrict__ C, int ldc,
                    const float* __restrict__ res, int ldres,
                    int M, int N, int act)
{
    __shared__ _Float16 sB[64][32];     // [n][k] panel for current k-step (4 KB)
    const int tid  = threadIdx.x;
    const int wave = tid >> 5;
    const int lane = tid & 31;
    const int bm = blockIdx.x * 64;
    const int bn = blockIdx.y * 64;
    const int kb = (lane >> 4) * 8;     // per-lane K sub-block (ISA 16-bit A/B layout)
    const int arow = bm + wave * 16 + (lane & 15);
    const _Float16* aptr = Ah + (size_t)arow * Kp;

    f32x8 acc[4];
    for (int i = 0; i < 4; i++) acc[i] = (f32x8){0.f,0.f,0.f,0.f,0.f,0.f,0.f,0.f};

    for (int k0 = 0; k0 < Kp; k0 += 32) {
        if (wave == 0) {
            tdm_load_tile((unsigned)(size_t)&sB[0][0],
                          Wt + (size_t)bn * Kp + k0, 32, 64, Kp);
            __builtin_amdgcn_s_wait_tensorcnt(0);
        }
        __syncthreads();
        if (k0 + 32 < Kp)   // prefetch next A fragment line -> global_prefetch_b8
            __builtin_prefetch(aptr + k0 + 32, 0, 1);

        f16x8 alo = *(const f16x8*)(aptr + k0 + kb);
        f16x8 ahi = *(const f16x8*)(aptr + k0 + 16 + kb);
        f16x16 af = __builtin_shufflevector(alo, ahi, 0,1,2,3,4,5,6,7,8,9,10,11,12,13,14,15);
#pragma unroll
        for (int nt = 0; nt < 4; nt++) {
            int nr = nt * 16 + (lane & 15);
            f16x8 blo = *(const f16x8*)&sB[nr][kb];
            f16x8 bhi = *(const f16x8*)&sB[nr][16 + kb];
            f16x16 bf = __builtin_shufflevector(blo, bhi, 0,1,2,3,4,5,6,7,8,9,10,11,12,13,14,15);
            acc[nt] = __builtin_amdgcn_wmma_f32_16x16x32_f16(
                          false, af, false, bf, (short)0, acc[nt], false, false);
        }
        __syncthreads();
    }

#pragma unroll
    for (int nt = 0; nt < 4; nt++) {
        int col = bn + nt * 16 + (lane & 15);
        float bc = (bias && col < N) ? bias[col] : 0.f;
        for (int r = 0; r < 8; ++r) {
            int row = bm + wave * 16 + ((lane >> 4) * 8) + r;  // C/D layout: lane-half => M+8
            if (row < M && col < N) {
                float v = act_apply(acc[nt][r] + bc, act);
                if (res) v += res[(size_t)row * ldres + col];
                C[(size_t)row * ldc + col] = v;
            }
        }
    }
}

// ---------------- farthest point sampling: 1 block per (b, blk) ----------------
__global__ void cmden_fps(const float* __restrict__ blocks, float* __restrict__ sampled, int m)
{
    int bj = blockIdx.x;
    const float* pts = blocks + (size_t)bj * PPBQ * 6;
    int b = bj / NBQ, j = bj % NBQ;
    float* out = sampled + ((size_t)b * (NBQ * m) + (size_t)j * m) * 6;
    __shared__ float d[PPBQ];
    __shared__ float rv[256];
    __shared__ int   ri[256];
    __shared__ float cur[3];
    int t = threadIdx.x;
    if (t == 0) { cur[0] = pts[0]; cur[1] = pts[1]; cur[2] = pts[2]; }
    if (t < 6) out[t] = pts[t];
    __syncthreads();
    for (int i = t; i < PPBQ; i += 256) {
        float dx = pts[i*6+0]-cur[0], dy = pts[i*6+1]-cur[1], dz = pts[i*6+2]-cur[2];
        d[i] = dx*dx + dy*dy + dz*dz;
    }
    __syncthreads();
    for (int s = 1; s < m; s++) {
        float bv = -1.f; int bi = 0;
        for (int i = t; i < PPBQ; i += 256)
            if (d[i] > bv) { bv = d[i]; bi = i; }
        rv[t] = bv; ri[t] = bi;
        __syncthreads();
        for (int off = 128; off > 0; off >>= 1) {
            if (t < off) {
                if (rv[t+off] > rv[t] || (rv[t+off] == rv[t] && ri[t+off] < ri[t])) {
                    rv[t] = rv[t+off]; ri[t] = ri[t+off];
                }
            }
            __syncthreads();
        }
        int nxt = ri[0];
        if (t == 0) { cur[0]=pts[nxt*6]; cur[1]=pts[nxt*6+1]; cur[2]=pts[nxt*6+2]; }
        if (t < 6) out[(size_t)s*6+t] = pts[nxt*6+t];
        __syncthreads();
        for (int i = t; i < PPBQ; i += 256) {
            float dx = pts[i*6+0]-cur[0], dy = pts[i*6+1]-cur[1], dz = pts[i*6+2]-cur[2];
            d[i] = fminf(d[i], dx*dx + dy*dy + dz*dz);
        }
        __syncthreads();
    }
}

// ---------------- pairwise squared distances ----------------
__global__ void cmden_pairdist(const float* __restrict__ s, float* __restrict__ d2, int N)
{
    long long t = (long long)blockIdx.x * blockDim.x + threadIdx.x;
    long long tot = (long long)BQ * N * N;
    if (t >= tot) return;
    int b = (int)(t / ((long long)N * N));
    int rem = (int)(t % ((long long)N * N));
    int i = rem / N, j = rem % N;
    const float* pi = s + ((size_t)b * N + i) * 6;
    const float* pj = s + ((size_t)b * N + j) * 6;
    float dx = pi[0]-pj[0], dy = pi[1]-pj[1], dz = pi[2]-pj[2];
    d2[t] = dx*dx + dy*dy + dz*dz;
}

// ---------------- KNN select (33 smallest, drop self) ----------------
__global__ void cmden_knnsel(const float* __restrict__ d2, int* __restrict__ knn, int N)
{
    int t = blockIdx.x * blockDim.x + threadIdx.x;
    if (t >= BQ * N) return;
    int b = t / N, i = t % N;
    const float* row = d2 + ((size_t)b * N + i) * N;
    float lastv = -1.f; int lasti = -1;
    for (int sel = 0; sel < KNNQ + 1; sel++) {
        float bv = 3.4e38f; int bi = N;
        for (int j = 0; j < N; j++) {
            float v = row[j];
            if (v < lastv || (v == lastv && j <= lasti)) continue;
            if (v < bv || (v == bv && j < bi)) { bv = v; bi = j; }
        }
        if (sel > 0) knn[((size_t)b * N + i) * KNNQ + (sel - 1)] = bi;
        lastv = bv; lasti = bi;
    }
}

// ---------------- build rel(xyz) and [nn|var] features ----------------
__global__ void cmden_rel(const float* __restrict__ s, const int* __restrict__ knn,
                          float* __restrict__ rel, float* __restrict__ nnvar, int N)
{
    long long t = (long long)blockIdx.x * blockDim.x + threadIdx.x;
    long long tot = (long long)BQ * N * KNNQ;
    if (t >= tot) return;
    int k = (int)(t % KNNQ);
    int bi = (int)(t / KNNQ);
    int b = bi / N, i = bi % N;
    int j = knn[((size_t)b * N + i) * KNNQ + k];
    const float* pi = s + ((size_t)b * N + i) * 6;
    const float* pj = s + ((size_t)b * N + j) * 6;
    rel[t*3+0] = pj[0]-pi[0]; rel[t*3+1] = pj[1]-pi[1]; rel[t*3+2] = pj[2]-pi[2];
    float dot = pi[3]*pj[3] + pi[4]*pj[4] + pi[5]*pj[5];
    nnvar[t*4+0] = pj[3]; nnvar[t*4+1] = pj[4]; nnvar[t*4+2] = pj[5];
    nnvar[t*4+3] = 1.f - dot;
}

// ---------------- max over 32 neighbors ----------------
__global__ void cmden_rowmax(const float* __restrict__ src, float* __restrict__ dst,
                             int coloff, int ldd, int rows)
{
    long long t = (long long)blockIdx.x * blockDim.x + threadIdx.x;
    if (t >= (long long)rows * HQ) return;
    int r = (int)(t / HQ), c = (int)(t % HQ);
    const float* p = src + ((size_t)r * KNNQ) * HQ + c;
    float m = p[0];
    for (int k = 1; k < KNNQ; k++) m = fmaxf(m, p[(size_t)k * HQ]);
    dst[(size_t)r * ldd + coloff + c] = m;
}

// ---------------- concat [points(6) | enh(128)] -> 134 ----------------
__global__ void cmden_catin(const float* __restrict__ s, const float* __restrict__ enh,
                            float* __restrict__ fin, int rows)
{
    long long t = (long long)blockIdx.x * blockDim.x + threadIdx.x;
    if (t >= (long long)rows * 134) return;
    int r = (int)(t / 134), j = (int)(t % 134);
    fin[t] = (j < 6) ? s[(size_t)r * 6 + j] : enh[(size_t)r * HQ + (j - 6)];
}

// ---------------- per-batch xyz min/max ----------------
__global__ void cmden_minmax(const float* __restrict__ s, float* __restrict__ mnmx, int N)
{
    __shared__ float red[256];
    int b = blockIdx.x, t = threadIdx.x;
    for (int ch = 0; ch < 3; ch++) {
        float mn = 3.4e38f, mx = -3.4e38f;
        for (int i = t; i < N; i += 256) {
            float v = s[((size_t)b * N + i) * 6 + ch];
            mn = fminf(mn, v); mx = fmaxf(mx, v);
        }
        red[t] = mn; __syncthreads();
        for (int o = 128; o > 0; o >>= 1) { if (t < o) red[t] = fminf(red[t], red[t+o]); __syncthreads(); }
        if (t == 0) mnmx[b*6 + ch] = red[0];
        __syncthreads();
        red[t] = mx; __syncthreads();
        for (int o = 128; o > 0; o >>= 1) { if (t < o) red[t] = fmaxf(red[t], red[t+o]); __syncthreads(); }
        if (t == 0) mnmx[b*6 + 3 + ch] = red[0];
        __syncthreads();
    }
}

// ---------------- serialization keys ----------------
__global__ void cmden_keys(const float* __restrict__ s, const float* __restrict__ mnmx,
                           int* __restrict__ keys, int N, int scheme)
{
    int t = blockIdx.x * blockDim.x + threadIdx.x;
    if (t >= BQ * N) return;
    int b = t / N, i = t % N;
    int c[3];
    for (int ch = 0; ch < 3; ch++) {
        float mn = mnmx[b*6+ch], mx = mnmx[b*6+3+ch];
        float v = s[((size_t)b * N + i) * 6 + ch];
        int q = (int)((v - mn) / (mx - mn + 1e-9f) * (float)RQ);
        c[ch] = min(max(q, 0), RQ - 1);
    }
    int key = 0;
    if (scheme == 2) {                       // raster
        key = (c[0] * RQ + c[1]) * RQ + c[2];
    } else if (scheme == 3) {                // zigzag
        int y = (c[0] & 1) ? (RQ-1-c[1]) : c[1];
        int z = ((c[0] + y) & 1) ? (RQ-1-c[2]) : c[2];
        key = (c[0] * RQ + y) * RQ + z;
    } else if (scheme == 1) {                // z_order
        for (int bb = 0; bb < BITSQ; bb++)
            for (int ii = 0; ii < 3; ii++)
                key |= ((c[ii] >> bb) & 1) << (3*bb + (2-ii));
    } else {                                 // hilbert
        int x[3] = {c[0], c[1], c[2]};
        for (int Q = 1 << (BITSQ-1); Q > 1; Q >>= 1) {
            int P = Q - 1;
            for (int ii = 0; ii < 3; ii++) {
                if (x[ii] & Q) x[0] ^= P;
                else { int tt = (x[0] ^ x[ii]) & P; x[0] ^= tt; x[ii] ^= tt; }
            }
        }
        x[1] ^= x[0]; x[2] ^= x[1];
        int tt = 0;
        for (int Q = 1 << (BITSQ-1); Q > 1; Q >>= 1) if (x[2] & Q) tt ^= Q - 1;
        for (int ii = 0; ii < 3; ii++) x[ii] ^= tt;
        for (int bb = 0; bb < BITSQ; bb++)
            for (int ii = 0; ii < 3; ii++)
                key |= ((x[ii] >> bb) & 1) << (3*bb + (2-ii));
    }
    keys[t] = key;
}

// ---------------- per-batch bitonic argsort (N = power of 2, <=1024) ----------------
__global__ void cmden_sort(const int* __restrict__ keys, int* __restrict__ order,
                           int* __restrict__ inv, int N)
{
    __shared__ int sk[1024];
    __shared__ int si[1024];
    int b = blockIdx.x, t = threadIdx.x;
    sk[t] = keys[b * N + t]; si[t] = t;
    __syncthreads();
    for (int size = 2; size <= N; size <<= 1) {
        for (int stride = size >> 1; stride > 0; stride >>= 1) {
            int p = t ^ stride;
            if (p > t) {
                bool up = ((t & size) == 0);
                bool gt = (sk[t] > sk[p]) || (sk[t] == sk[p] && si[t] > si[p]);
                if (up == gt) {
                    int tk = sk[t]; sk[t] = sk[p]; sk[p] = tk;
                    int ti = si[t]; si[t] = si[p]; si[p] = ti;
                }
            }
            __syncthreads();
        }
    }
    order[b * N + t] = si[t];
    inv[b * N + si[t]] = t;
}

// ---------------- seq = [emb ; feats[order] ; emb] ----------------
__global__ void cmden_gatherseq(const float* __restrict__ feats, const int* __restrict__ order,
                                const float* __restrict__ emb, float* __restrict__ seq,
                                int N, int L)
{
    long long t = (long long)blockIdx.x * blockDim.x + threadIdx.x;
    if (t >= (long long)BQ * L * HQ) return;
    int c = (int)(t % HQ);
    int bl = (int)(t / HQ);
    int b = bl / L, l = bl % L;
    float v;
    if (l == 0 || l == L - 1) v = emb[c];
    else v = feats[((size_t)b * N + order[b * N + (l-1)]) * HQ + c];
    seq[t] = v;
}

// ---------------- row layernorm (1 block per row) ----------------
__global__ void cmden_layernorm(const float* __restrict__ x, const float* __restrict__ w,
                                const float* __restrict__ bb, float* __restrict__ y, int D)
{
    __shared__ float red[256];
    int row = blockIdx.x, t = threadIdx.x, T = blockDim.x;
    const float* xr = x + (size_t)row * D;
    float s = 0.f;
    for (int i = t; i < D; i += T) s += xr[i];
    red[t] = s; __syncthreads();
    for (int o = T >> 1; o > 0; o >>= 1) { if (t < o) red[t] += red[t+o]; __syncthreads(); }
    float mean = red[0] / D;
    __syncthreads();
    float v = 0.f;
    for (int i = t; i < D; i += T) { float dd = xr[i] - mean; v += dd*dd; }
    red[t] = v; __syncthreads();
    for (int o = T >> 1; o > 0; o >>= 1) { if (t < o) red[t] += red[t+o]; __syncthreads(); }
    float inv = rsqrtf(red[0] / D + 1e-5f);
    for (int i = t; i < D; i += T)
        y[(size_t)row * D + i] = (xr[i] - mean) * inv * w[i] + bb[i];
}

// ---------------- causal depthwise conv (K=4) + silu ----------------
__global__ void cmden_conv(const float* __restrict__ xz, const float* __restrict__ cw,
                           const float* __restrict__ cb, float* __restrict__ xcs, int L)
{
    long long t = (long long)blockIdx.x * blockDim.x + threadIdx.x;
    if (t >= (long long)BQ * L * DIQ) return;
    int d = (int)(t % DIQ);
    int bl = (int)(t / DIQ);
    int b = bl / L, l = bl % L;
    float acc = cb[d];
    for (int k = 0; k < KCQ; k++) {
        int lp = l - (KCQ - 1) + k;
        if (lp >= 0) acc += xz[((size_t)b * L + lp) * (2*DIQ) + d] * cw[d * KCQ + k];
    }
    xcs[t] = acc / (1.f + __expf(-acc));
}

// ---------------- selective scan (1 block per batch, thread = channel d) ----------------
__global__ void cmden_scan(const float* __restrict__ dt, const float* __restrict__ proj,
                           const float* __restrict__ xcs, const float* __restrict__ xz,
                           const float* __restrict__ A_log, const float* __restrict__ Dp,
                           float* __restrict__ ym, int L)
{
    __shared__ float sB[DSQ], sC[DSQ];
    int b = blockIdx.x, d = threadIdx.x;
    float hst[DSQ], Ad[DSQ];
    for (int s = 0; s < DSQ; s++) { hst[s] = 0.f; Ad[s] = -__expf(A_log[d * DSQ + s]); }
    float Dd = Dp[d];
    for (int l = 0; l < L; l++) {
        if (d < DSQ) {
            sB[d] = proj[((size_t)b * L + l) * 40 + DTRQ + d];
            sC[d] = proj[((size_t)b * L + l) * 40 + DTRQ + DSQ + d];
        }
        __syncthreads();
        float dtv = dt[((size_t)b * L + l) * DIQ + d];
        float xcv = xcs[((size_t)b * L + l) * DIQ + d];
        float y = 0.f;
        for (int s = 0; s < DSQ; s++) {
            hst[s] = __expf(dtv * Ad[s]) * hst[s] + dtv * sB[s] * xcv;
            y += hst[s] * sC[s];
        }
        y += xcv * Dd;
        float zv = xz[((size_t)b * L + l) * (2*DIQ) + DIQ + d];
        ym[((size_t)b * L + l) * DIQ + d] = y * (zv / (1.f + __expf(-zv)));
        __syncthreads();
    }
}

// ---------------- feats[b,i] = seq_out[b, inv[i]+1] ----------------
__global__ void cmden_ungather(const float* __restrict__ seqo, const int* __restrict__ inv,
                               float* __restrict__ feats, int N, int L)
{
    long long t = (long long)blockIdx.x * blockDim.x + threadIdx.x;
    if (t >= (long long)BQ * N * HQ) return;
    int c = (int)(t % HQ);
    int bi = (int)(t / HQ);
    int b = bi / N, i = bi % N;
    feats[t] = seqo[((size_t)b * L + inv[b * N + i] + 1) * HQ + c];
}

// ---------------- pooled = [max ; mean] over N ----------------
__global__ void cmden_pool(const float* __restrict__ feats, float* __restrict__ pooled, int N)
{
    int b = blockIdx.x, c = threadIdx.x;  // 128 threads
    float mx = -3.4e38f, s = 0.f;
    for (int i = 0; i < N; i++) {
        float v = feats[((size_t)b * N + i) * HQ + c];
        mx = fmaxf(mx, v); s += v;
    }
    pooled[b * (2*HQ) + c] = mx;
    pooled[b * (2*HQ) + HQ + c] = s / (float)N;
}

// ---------------- out = prev[i/2] + dec ----------------
__global__ void cmden_expandadd(const float* __restrict__ prev, const float* __restrict__ dec,
                                float* __restrict__ out, int n, int pn)
{
    int t = blockIdx.x * blockDim.x + threadIdx.x;
    if (t >= BQ * n * 3) return;
    int b = t / (n * 3), r = t % (n * 3);
    int i = r / 3, c = r % 3;
    out[t] = prev[(size_t)b * pn * 3 + (i >> 1) * 3 + c] + dec[(size_t)b * n * 3 + r];
}

__global__ void cmden_copy(const float* __restrict__ src, float* __restrict__ dst, int n)
{
    int t = blockIdx.x * blockDim.x + threadIdx.x;
    if (t < n) dst[t] = src[t];
}

// ============================ host orchestration ============================
extern "C" void kernel_launch(void* const* d_in, const int* in_sizes, int n_in,
                              void* d_out, int out_size, void* d_ws, size_t ws_size,
                              hipStream_t stream)
{
    (void)in_sizes; (void)n_in; (void)out_size; (void)ws_size;
    auto P = [&](int i) -> const float* { return (const float*)d_in[i]; };
    const float* arch_blocks = P(1);

    // param leaf indices (alphabetical pytree flatten):
    // 2..7 dec{high_b,high_w,low_b,low_w,mid_b,mid_w}
    // 8..19 enh{coord,normal,out}(w1,b1,w2,b2)  20..31 g1,g2,g3
    // 32 high / 93 low / 154 mid enc_p
    enum { AL=0, Dv=1, CB=2, CW=3, DTB=4, DTW=5, INB=6, INW=7, LNB=8, LNW=9, OUTB=10, OUTW=11, XPW=12 };

    // workspace arena
    char* wsp = (char*)d_ws;
    auto alloc = [&](size_t nf) -> float* {
        float* p = (float*)wsp;
        wsp += ((nf * sizeof(float) + 255) / 256) * 256;
        return p;
    };
    const int NMAX = 1024, LMAX = NMAX + 2;
    float* sampled = alloc((size_t)BQ * NMAX * 6);
    float* d2      = alloc((size_t)BQ * NMAX * NMAX);
    int*   knn     = (int*)alloc((size_t)BQ * NMAX * KNNQ);
    float* rel     = alloc((size_t)BQ * NMAX * KNNQ * 3);
    float* nnvar   = alloc((size_t)BQ * NMAX * KNNQ * 4);
    float* h1      = alloc((size_t)BQ * NMAX * KNNQ * HQ);
    float* h2      = alloc((size_t)BQ * NMAX * KNNQ * HQ);
    float* cfnf    = alloc((size_t)BQ * NMAX * 2 * HQ);
    float* enh     = alloc((size_t)BQ * NMAX * HQ);
    float* fin     = alloc((size_t)BQ * NMAX * 134);
    float* featsA  = alloc((size_t)BQ * NMAX * HQ);
    float* featsB  = alloc((size_t)BQ * NMAX * HQ);
    float* mnmx    = alloc(BQ * 6);
    int*   keysb   = (int*)alloc((size_t)BQ * NMAX);
    int*   order   = (int*)alloc((size_t)BQ * NMAX);
    int*   invp    = (int*)alloc((size_t)BQ * NMAX);
    float* seqx    = alloc((size_t)BQ * LMAX * HQ);
    float* hln     = alloc((size_t)BQ * LMAX * HQ);
    float* xz      = alloc((size_t)BQ * LMAX * 2 * DIQ);
    float* xcs     = alloc((size_t)BQ * LMAX * DIQ);
    float* proj    = alloc((size_t)BQ * LMAX * 40);
    float* dtb     = alloc((size_t)BQ * LMAX * DIQ);
    float* ym      = alloc((size_t)BQ * LMAX * DIQ);
    float* seqo    = alloc((size_t)BQ * LMAX * HQ);
    float* pooled  = alloc(BQ * 2 * HQ);
    float* pldn    = alloc(BQ * 2 * HQ);
    float* o1h     = alloc(BQ * GQ);
    float* gbuf    = alloc(BQ * 3 * GQ);
    float* ghid    = alloc(BQ * GQ);
    float* f1b     = alloc(BQ * GQ);
    float* f2b     = alloc(BQ * GQ);
    float* f3b     = alloc(BQ * GQ);
    float* deca    = alloc((size_t)BQ * 1024 * 3);
    // f16 staging arenas (pre-converted, padded operands for the WMMA GEMM)
    _Float16* AhBuf = (_Float16*)alloc(9 * 1024 * 1024);   // 18M halves
    _Float16* WtBuf = (_Float16*)alloc(1 * 1024 * 1024);   //  2M halves

    auto g1d = [](long long n, int b) { return (unsigned)((n + b - 1) / b); };

    auto gemm = [&](const float* A, int lda, const float* W, const float* bias,
                    float* C, int ldc, const float* res, int ldres,
                    int M, int N, int K, int act) {
        int Kp = (K + 31) & ~31;
        int Mp = (M + 63) & ~63;
        int Np = (N + 63) & ~63;
        cmden_cvtA<<<g1d((long long)Mp * Kp, 256), 256, 0, stream>>>(A, lda, AhBuf, M, K, Kp);
        cmden_cvtW<<<g1d((long long)Np * Kp, 256), 256, 0, stream>>>(W, WtBuf, K, N, Kp, Np);
        dim3 g(Mp / 64, Np / 64);
        cmden_gemm_f16<<<g, 128, 0, stream>>>(AhBuf, Kp, WtBuf, bias, C, ldc, res, ldres, M, N, act);
    };

    struct ScaleCfg { int N; int enc; int gcol; };
    const ScaleCfg scales[3] = { {256, 93, 0}, {512, 154, GQ}, {1024, 32, 2 * GQ} };

    for (int sidx = 0; sidx < 3; sidx++) {
        const int N = scales[sidx].N, E = scales[sidx].enc;
        const int m = N / NBQ, L = N + 2, MN = BQ * N;

        cmden_fps<<<BQ * NBQ, 256, 0, stream>>>(arch_blocks, sampled, m);
        cmden_pairdist<<<g1d((long long)BQ*N*N, 256), 256, 0, stream>>>(sampled, d2, N);
        cmden_knnsel<<<g1d(MN, 64), 64, 0, stream>>>(d2, knn, N);
        cmden_rel<<<g1d((long long)MN*KNNQ, 256), 256, 0, stream>>>(sampled, knn, rel, nnvar, N);

        // enhancer: coord path
        gemm(rel,   3,  P(8),  P(9),  h1, HQ, nullptr, 0, MN*KNNQ, HQ, 3,  1);
        gemm(h1,   HQ,  P(10), P(11), h2, HQ, nullptr, 0, MN*KNNQ, HQ, HQ, 0);
        cmden_rowmax<<<g1d((long long)MN*HQ, 256), 256, 0, stream>>>(h2, cfnf, 0, 2*HQ, MN);
        // normal path
        gemm(nnvar, 4,  P(12), P(13), h1, HQ, nullptr, 0, MN*KNNQ, HQ, 4,  1);
        gemm(h1,   HQ,  P(14), P(15), h2, HQ, nullptr, 0, MN*KNNQ, HQ, HQ, 0);
        cmden_rowmax<<<g1d((long long)MN*HQ, 256), 256, 0, stream>>>(h2, cfnf, HQ, 2*HQ, MN);
        // out mlp
        gemm(cfnf, 2*HQ, P(16), P(17), h1,  HQ, nullptr, 0, MN, HQ, 2*HQ, 1);
        gemm(h1,   HQ,   P(18), P(19), enh, HQ, nullptr, 0, MN, HQ, HQ,   0);

        // encoder input projection
        cmden_catin<<<g1d((long long)MN*134, 256), 256, 0, stream>>>(sampled, enh, fin, MN);
        gemm(fin, 134, P(E+54), P(E+53), featsA, HQ, nullptr, 0, MN, HQ, 134, 0);

        cmden_minmax<<<BQ, 256, 0, stream>>>(sampled, mnmx, N);

        float* cur = featsA; float* nxt = featsB;
        for (int sch = 0; sch < 4; sch++) {
            const int BL = E + sch * 13;
            cmden_keys<<<g1d(MN, 256), 256, 0, stream>>>(sampled, mnmx, keysb, N, sch);
            cmden_sort<<<BQ, N, 0, stream>>>(keysb, order, invp, N);
            cmden_gatherseq<<<g1d((long long)BQ*L*HQ, 256), 256, 0, stream>>>(
                cur, order, P(E+52) + sch * HQ, seqx, N, L);
            cmden_layernorm<<<BQ*L, 128, 0, stream>>>(seqx, P(BL+LNW), P(BL+LNB), hln, HQ);
            gemm(hln, HQ, P(BL+INW), P(BL+INB), xz, 2*DIQ, nullptr, 0, BQ*L, 2*DIQ, HQ, 0);
            cmden_conv<<<g1d((long long)BQ*L*DIQ, 256), 256, 0, stream>>>(xz, P(BL+CW), P(BL+CB), xcs, L);
            gemm(xcs, DIQ, P(BL+XPW), nullptr, proj, 40, nullptr, 0, BQ*L, 40, DIQ, 0);
            gemm(proj, 40, P(BL+DTW), P(BL+DTB), dtb, DIQ, nullptr, 0, BQ*L, DIQ, DTRQ, 3);
            cmden_scan<<<BQ, DIQ, 0, stream>>>(dtb, proj, xcs, xz, P(BL+AL), P(BL+Dv), ym, L);
            gemm(ym, DIQ, P(BL+OUTW), P(BL+OUTB), seqo, HQ, seqx, HQ, BQ*L, HQ, DIQ, 0);
            cmden_ungather<<<g1d((long long)MN*HQ, 256), 256, 0, stream>>>(seqo, invp, nxt, N, L);
            float* tmp = cur; cur = nxt; nxt = tmp;
        }

        cmden_pool<<<BQ, HQ, 0, stream>>>(cur, pooled, N);
        cmden_layernorm<<<BQ, 256, 0, stream>>>(pooled, P(E+56), P(E+55), pldn, 2*HQ);
        gemm(pldn, 2*HQ, P(E+58), P(E+57), o1h, GQ, nullptr, 0, BQ, GQ, 2*HQ, 1);
        gemm(o1h, GQ, P(E+60), P(E+59), gbuf + scales[sidx].gcol, 3*GQ, nullptr, 0, BQ, GQ, GQ, 0);
    }

    // global heads
    gemm(gbuf, 3*GQ, P(20), P(21), ghid, GQ, nullptr, 0, BQ, GQ, 3*GQ, 1);
    gemm(ghid, GQ,   P(22), P(23), f1b,  GQ, nullptr, 0, BQ, GQ, GQ,   0);
    gemm(gbuf, 3*GQ, P(24), P(25), ghid, GQ, nullptr, 0, BQ, GQ, 3*GQ, 1);
    gemm(ghid, GQ,   P(26), P(27), f2b,  GQ, nullptr, 0, BQ, GQ, GQ,   0);
    gemm(gbuf, 3*GQ, P(28), P(29), ghid, GQ, nullptr, 0, BQ, GQ, 3*GQ, 1);
    gemm(ghid, GQ,   P(30), P(31), f3b,  GQ, nullptr, 0, BQ, GQ, GQ,   0);

    // decoders -> output  [pl | pm | ph | g]
    float* out = (float*)d_out;
    float* pl = out;
    float* pm = out + BQ * 256 * 3;
    float* ph = pm + BQ * 512 * 3;
    float* gout = ph + BQ * 1024 * 3;

    gemm(f3b, GQ, P(5), P(4), pl, 256*3, nullptr, 0, BQ, 256*3, GQ, 0);
    gemm(f2b, GQ, P(7), P(6), deca, 512*3, nullptr, 0, BQ, 512*3, GQ, 0);
    cmden_expandadd<<<g1d(BQ*512*3, 256), 256, 0, stream>>>(pl, deca, pm, 512, 256);
    gemm(f1b, GQ, P(3), P(2), deca, 1024*3, nullptr, 0, BQ, 1024*3, GQ, 0);
    cmden_expandadd<<<g1d(BQ*1024*3, 256), 256, 0, stream>>>(pm, deca, ph, 1024, 512);
    cmden_copy<<<g1d(BQ*3*GQ, 256), 256, 0, stream>>>(gbuf, gout, BQ*3*GQ);
}
